// GraphSAGEBackbone_4578435137604
// MI455X (gfx1250) — compile-verified, matrix operand order checked by reading, not run on hardware
//
#include <hip/hip_runtime.h>
#include <hip/hip_bf16.h>

typedef __attribute__((ext_vector_type(16))) _Float16 v16h;
typedef __attribute__((ext_vector_type(8)))  _Float16 v8h;
typedef __attribute__((ext_vector_type(4)))  _Float16 v4h;
typedef __attribute__((ext_vector_type(8)))  float    v8f;

#define FD 128          // feature dim
#define LDS_PAD 144     // 128 + 16 halfs: 288B row stride, 16B-aligned vector stores

// ---------------------------------------------------------------- utilities
__global__ void zero_f32(float* __restrict__ p, int n) {
    int i = blockIdx.x * blockDim.x + threadIdx.x;
    if (i < n) p[i] = 0.0f;
}

__global__ void deg_kernel(const int* __restrict__ dst, float* __restrict__ deg, int E) {
    int e = blockIdx.x * blockDim.x + threadIdx.x;
    if (e < E) atomicAdd(&deg[dst[e]], 1.0f);
}

__global__ void invdeg_kernel(const float* __restrict__ deg, float* __restrict__ inv, int N) {
    int i = blockIdx.x * blockDim.x + threadIdx.x;
    if (i < N) {
        float d = deg[i];
        inv[i] = (d > 0.0f) ? (1.0f / d) : 0.0f;
    }
}

__global__ void convert_f32_to_f16(const float* __restrict__ w, _Float16* __restrict__ o, long n) {
    long i = (long)blockIdx.x * blockDim.x + threadIdx.x;
    if (i < n) o[i] = (_Float16)w[i];
}

// ------------------------------------------------- edge scatter (mean numerator)
// one wave (32 lanes) per edge; h lives in f16 -> 256B random gather per edge
// (half the bytes of f32 on the bandwidth-critical stream); accumulate in f32.
__global__ void __launch_bounds__(256)
scatter_kernel(const _Float16* __restrict__ h16, const int* __restrict__ src,
               const int* __restrict__ dst, float* __restrict__ agg, int E) {
    int wv   = (blockIdx.x * blockDim.x + threadIdx.x) >> 5;
    int lane = threadIdx.x & 31;
    if (wv >= E) return;
    int s = src[wv];
    int d = dst[wv];
    v4h v = ((const v4h*)(h16 + (size_t)s * FD))[lane];   // 8B per lane
    float* ap = agg + (size_t)d * FD + lane * 4;
    atomicAdd(ap + 0, (float)v.x);
    atomicAdd(ap + 1, (float)v.y);
    atomicAdd(ap + 2, (float)v.z);
    atomicAdd(ap + 3, (float)v.w);
}

// ------------------------------------------------- fused dual-GEMM + BN stats
// out = (agg*invdeg) @ Wl^T + h @ Wr^T + bl ; accumulates per-column sum/sumsq.
// Block = 256 threads (8 waves). Block owns rows [row0, row0+16); wave w owns
// output columns [16w, 16w+16). K = 128 -> 4 wmma steps per operand pair.
__global__ void __launch_bounds__(256)
gemm_bn_stats(const float* __restrict__ agg, const _Float16* __restrict__ h16,
              const float* __restrict__ invdeg,
              const _Float16* __restrict__ WlH,   // [128][128] f16, row n = output col
              const _Float16* __restrict__ WrH,
              const float* __restrict__ bias,     // [128]
              float* __restrict__ outp,
              float* __restrict__ sums, float* __restrict__ sumsq,
              int N) {
    __shared__ _Float16 sAgg[16][LDS_PAD];
    __shared__ _Float16 sH[16][LDS_PAD];

    const int tid  = threadIdx.x;
    const int row0 = blockIdx.x * 16;

    // stage A tiles: 2048 halfs each, 256 threads -> 8 halfs (16B) per thread.
    {
        const int r = tid >> 4;            // 0..15 row in tile
        const int k = (tid & 15) * 8;      // 0..120 start column
        const int row = row0 + r;
        // h tile: raw b128 copy (h already f16)
        uint4 hv = {0u, 0u, 0u, 0u};
        if (row < N) hv = *(const uint4*)(h16 + (size_t)row * FD + k);
        *(uint4*)&sH[r][k] = hv;
        // agg tile: f32 load, normalize by inv_deg, pack to f16
        v8h ap = (v8h)(_Float16)0;
        if (row < N) {
            const float idg = invdeg[row];
            const float4 a0 = *(const float4*)(agg + (size_t)row * FD + k);
            const float4 a1 = *(const float4*)(agg + (size_t)row * FD + k + 4);
            ap.s0 = (_Float16)(a0.x * idg); ap.s1 = (_Float16)(a0.y * idg);
            ap.s2 = (_Float16)(a0.z * idg); ap.s3 = (_Float16)(a0.w * idg);
            ap.s4 = (_Float16)(a1.x * idg); ap.s5 = (_Float16)(a1.y * idg);
            ap.s6 = (_Float16)(a1.z * idg); ap.s7 = (_Float16)(a1.w * idg);
        }
        *(v8h*)&sAgg[r][k] = ap;
    }
    __syncthreads();

    const int lane = tid & 31;
    const int hl   = lane & 15;       // row-of-A / col-of-B held by this lane
    const int hi   = lane >> 4;       // half-wave selector
    const int nt   = tid >> 5;        // wave id == column tile
    const int n    = nt * 16 + hl;    // output column owned by this lane

    v8f c = {};

    // ---- agg @ Wl^T
    #pragma unroll
    for (int kc = 0; kc < 4; ++kc) {
        const int k0 = kc * 32 + hi * 8;
        v16h a, b;
        #pragma unroll
        for (int v = 0; v < 4; ++v) {
            int k  = k0 + 2 * v;        // fragment slots 2v, 2v+1
            int k2 = k + 16;            // fragment slots 2v+8, 2v+9
            a[2*v]     = sAgg[hl][k];
            a[2*v + 1] = sAgg[hl][k + 1];
            a[2*v + 8] = sAgg[hl][k2];
            a[2*v + 9] = sAgg[hl][k2 + 1];
            b[2*v]     = WlH[(size_t)n * FD + k];
            b[2*v + 1] = WlH[(size_t)n * FD + k + 1];
            b[2*v + 8] = WlH[(size_t)n * FD + k2];
            b[2*v + 9] = WlH[(size_t)n * FD + k2 + 1];
        }
        c = __builtin_amdgcn_wmma_f32_16x16x32_f16(false, a, false, b,
                                                   (short)0, c, false, false);
    }

    // ---- h @ Wr^T (accumulate into same C)
    #pragma unroll
    for (int kc = 0; kc < 4; ++kc) {
        const int k0 = kc * 32 + hi * 8;
        v16h a, b;
        #pragma unroll
        for (int v = 0; v < 4; ++v) {
            int k  = k0 + 2 * v;
            int k2 = k + 16;
            a[2*v]     = sH[hl][k];
            a[2*v + 1] = sH[hl][k + 1];
            a[2*v + 8] = sH[hl][k2];
            a[2*v + 9] = sH[hl][k2 + 1];
            b[2*v]     = WrH[(size_t)n * FD + k];
            b[2*v + 1] = WrH[(size_t)n * FD + k + 1];
            b[2*v + 8] = WrH[(size_t)n * FD + k2];
            b[2*v + 9] = WrH[(size_t)n * FD + k2 + 1];
        }
        c = __builtin_amdgcn_wmma_f32_16x16x32_f16(false, a, false, b,
                                                   (short)0, c, false, false);
    }

    // ---- epilogue: bias, store pre-BN activations, per-column partial stats
    const float bn = bias[n];
    float s = 0.0f, s2 = 0.0f;
    #pragma unroll
    for (int g = 0; g < 8; ++g) {
        int m = row0 + g + 8 * hi;    // C layout: vgpr g, half-wave hi -> row
        if (m < N) {
            float val = c[g] + bn;
            outp[(size_t)m * FD + n] = val;
            s  += val;
            s2 += val * val;
        }
    }
    atomicAdd(&sums[n],  s);
    atomicAdd(&sumsq[n], s2);
}

// ------------------------------------------------- BN finalize (1 block / 128 thr)
__global__ void bn_finalize(const float* __restrict__ sums, const float* __restrict__ sumsq,
                            const float* __restrict__ gamma, const float* __restrict__ beta,
                            float* __restrict__ scale, float* __restrict__ shift,
                            float invN) {
    int d = threadIdx.x;
    if (d < FD) {
        float mu  = sums[d] * invN;
        float var = sumsq[d] * invN - mu * mu;
        float sc  = rsqrtf(var + 1e-5f) * gamma[d];
        scale[d] = sc;
        shift[d] = beta[d] - mu * sc;
    }
}

// ------------------------------------------------- BN apply + ReLU
// Intermediate layers: write f16 h for the next layer's gather/GEMM.
// Final layer: write f32 to d_out (reference output dtype).
__global__ void bn_apply_relu(const float* __restrict__ outp,
                              const float* __restrict__ scale,
                              const float* __restrict__ shift,
                              _Float16* __restrict__ dst16,
                              float* __restrict__ dst32, long n) {
    long i = (long)blockIdx.x * blockDim.x + threadIdx.x;
    if (i < n) {
        int d = (int)(i & (FD - 1));
        float v = outp[i] * scale[d] + shift[d];
        v = (v > 0.0f) ? v : 0.0f;
        if (dst32) dst32[i] = v;
        else       dst16[i] = (_Float16)v;
    }
}

// =========================================================== host-side launcher
extern "C" void kernel_launch(void* const* d_in, const int* in_sizes, int n_in,
                              void* d_out, int out_size, void* d_ws, size_t ws_size,
                              hipStream_t stream) {
    const float* x     = (const float*)d_in[0];
    const int*   ei    = (const int*)  d_in[1];
    const float* Wl    = (const float*)d_in[2];
    const float* bl    = (const float*)d_in[3];
    const float* Wr    = (const float*)d_in[4];
    const float* gamma = (const float*)d_in[5];
    const float* beta  = (const float*)d_in[6];

    const int N = in_sizes[0] / FD;       // 50000
    const int E = in_sizes[1] / 2;        // 1600000
    const int L = in_sizes[3] / FD;       // 3
    const size_t ND = (size_t)N * FD;

    const int* srcI = ei;
    const int* dstI = ei + E;

    // ---- carve workspace
    char* w = (char*)d_ws;
    float* deg    = (float*)w; w += sizeof(float) * (size_t)N;
    float* invd   = (float*)w; w += sizeof(float) * (size_t)N;
    float* agg    = (float*)w; w += sizeof(float) * ND;
    float* outp   = (float*)w; w += sizeof(float) * ND;
    float* sums   = (float*)w; w += sizeof(float) * FD;
    float* sumsq  = (float*)w; w += sizeof(float) * FD;
    float* scale  = (float*)w; w += sizeof(float) * FD;
    float* shift  = (float*)w; w += sizeof(float) * FD;
    _Float16* h16 = (_Float16*)w; w += sizeof(_Float16) * ND;
    _Float16* WlH = (_Float16*)w; w += sizeof(_Float16) * (size_t)L * FD * FD;
    _Float16* WrH = (_Float16*)w;

    // ---- h0 = f16(x)
    convert_f32_to_f16<<<(int)((ND + 255) / 256), 256, 0, stream>>>(x, h16, (long)ND);

    // ---- degrees -> inv_deg
    zero_f32<<<(N + 255) / 256, 256, 0, stream>>>(deg, N);
    deg_kernel<<<(E + 255) / 256, 256, 0, stream>>>(dstI, deg, E);
    invdeg_kernel<<<(N + 255) / 256, 256, 0, stream>>>(deg, invd, N);

    // ---- convert all layer weights to f16 once
    const long WN = (long)L * FD * FD;
    convert_f32_to_f16<<<(int)((WN + 255) / 256), 256, 0, stream>>>(Wl, WlH, WN);
    convert_f32_to_f16<<<(int)((WN + 255) / 256), 256, 0, stream>>>(Wr, WrH, WN);

    const int rowTiles  = (N + 15) / 16;                 // 3125
    const int elemBlks  = (int)((ND + 255) / 256);       // 25000
    const int scatBlks  = (E + 7) / 8;                   // 8 waves/block

    for (int l = 0; l < L; ++l) {
        // zero aggregation buffer + BN stat accumulators (contiguous 256 floats)
        zero_f32<<<elemBlks, 256, 0, stream>>>(agg, (int)ND);
        zero_f32<<<1, 256, 0, stream>>>(sums, 2 * FD);

        scatter_kernel<<<scatBlks, 256, 0, stream>>>(h16, srcI, dstI, agg, E);

        gemm_bn_stats<<<rowTiles, 256, 0, stream>>>(
            agg, h16, invd,
            WlH + (size_t)l * FD * FD, WrH + (size_t)l * FD * FD,
            bl + (size_t)l * FD,
            outp, sums, sumsq, N);

        bn_finalize<<<1, 128, 0, stream>>>(sums, sumsq,
                                           gamma + (size_t)l * FD, beta + (size_t)l * FD,
                                           scale, shift, 1.0f / (float)N);

        const bool last = (l == L - 1);
        bn_apply_relu<<<elemBlks, 256, 0, stream>>>(outp, scale, shift,
                                                    last ? (_Float16*)nullptr : h16,
                                                    last ? (float*)d_out : nullptr,
                                                    (long)ND);
    }
}
